// CenterReppHead_68616397521179
// MI455X (gfx1250) — matrix operand order repaired and checked by reading, not compiled, not run on hardware
//
#include <hip/hip_runtime.h>
#include <hip/hip_bf16.h>

typedef __bf16 bf16_t;
typedef __attribute__((ext_vector_type(16))) __bf16 v16bf;
typedef __attribute__((ext_vector_type(8)))  __bf16 v8bf;
typedef __attribute__((ext_vector_type(8)))  float  v8f;
typedef __attribute__((ext_vector_type(4)))  unsigned int v4u;

#define BB   2
#define CC   256
#define HH   128
#define WW   128
#define PW   130                 // padded width  (1-px halo each side)
#define PH   130                 // padded height
#define HWP  (HH*WW)             // 16384
#define PTOT (BB*HWP)            // 32768 pixels (unpadded)
#define PIXP (PH*PW)             // 16900 padded pixels per batch
#define KSTEPS3 72               // 9*256/32
#define KSTEPS1 8                // 256/32
#define MTILES  (PTOT/16)        // 2048

static __device__ __forceinline__ v8f wmma_bf16(v16bf a, v16bf b, v8f c) {
  // D = A(16x32 bf16) * B(32x16 bf16) + C(16x16 f32)
  return __builtin_amdgcn_wmma_f32_16x16x32_bf16(false, a, false, b, (short)0, c, false, false);
}

// padded NHWC pixel base (element index) for unpadded coords (y,x) in [-1,128]
static __device__ __forceinline__ size_t ppix(int b, int y, int x) {
  return (((size_t)b * PH + (y + 1)) * PW + (x + 1)) << 8;   // *256 channels
}

// ---------------------------------------------------------------------------
// Bulk zero (halo init for padded activation buffers) — 16B per thread
// ---------------------------------------------------------------------------
__global__ void k_zero16(v4u* __restrict__ p, size_t n16) {
  size_t i = (size_t)blockIdx.x * blockDim.x + threadIdx.x;
  if (i < n16) { v4u z = {0u, 0u, 0u, 0u}; p[i] = z; }
}

// NCHW f32 -> padded NHWC bf16 (interior only)
__global__ void k_nchw_to_nhwc(const float* __restrict__ in, bf16_t* __restrict__ out) {
  size_t idx = (size_t)blockIdx.x * blockDim.x + threadIdx.x;
  if (idx >= (size_t)PTOT * CC) return;
  int x = idx % WW;
  int y = (idx / WW) % HH;
  int c = (idx / HWP) % CC;
  int b = idx / ((size_t)HWP * CC);
  out[ppix(b, y, x) + c] = (bf16_t)in[idx];
}

// OIHW (256,256,3,3) f32 -> packed B-fragments [nt=16][kstep=72][lane=32][e=16]
__global__ void k_pack_w3x3(const float* __restrict__ w, bf16_t* __restrict__ out) {
  size_t idx = (size_t)blockIdx.x * blockDim.x + threadIdx.x;
  if (idx >= (size_t)16 * KSTEPS3 * 32 * 16) return;
  int e     = idx & 15;
  int lane  = (idx >> 4) & 31;
  int kstep = (idx >> 9) % KSTEPS3;
  int nt    = (idx >> 9) / KSTEPS3;
  int k     = kstep * 32 + ((e >> 3) << 4) + ((lane >> 4) << 3) + (e & 7);
  int kpos  = k >> 8;
  int ic    = k & 255;
  int oc    = nt * 16 + (lane & 15);
  out[idx] = (bf16_t)w[((size_t)oc * CC + ic) * 9 + kpos];
}

// (O,256,1,1) f32 -> packed [nt][8][32][16], zero-padded past O
__global__ void k_pack_w1x1(const float* __restrict__ w, bf16_t* __restrict__ out,
                            int O, int ntiles) {
  size_t idx = (size_t)blockIdx.x * blockDim.x + threadIdx.x;
  if (idx >= (size_t)ntiles * KSTEPS1 * 32 * 16) return;
  int e     = idx & 15;
  int lane  = (idx >> 4) & 31;
  int kstep = (idx >> 9) % KSTEPS1;
  int nt    = (idx >> 9) / KSTEPS1;
  int ic    = kstep * 32 + ((e >> 3) << 4) + ((lane >> 4) << 3) + (e & 7);
  int oc    = nt * 16 + (lane & 15);
  float v = (oc < O) ? w[(size_t)oc * CC + ic] : 0.0f;
  out[idx] = (bf16_t)v;
}

// ---------------------------------------------------------------------------
// 3x3 conv implicit GEMM, halo-padded input, register-double-buffered.
// block = 128 (4 waves); wave w -> out-channels [w*64, w*64+64); grid = 2048
// ---------------------------------------------------------------------------
__global__ __launch_bounds__(128) void k_conv3x3(const bf16_t* __restrict__ act,
                                                 const bf16_t* __restrict__ wpack,
                                                 const float*  __restrict__ bias,
                                                 bf16_t* __restrict__ out) {
  const int lane = threadIdx.x & 31;
  const int wv   = threadIdx.x >> 5;
  const int tile = blockIdx.x;
  const int b    = tile >> 10;
  const int y    = (tile >> 3) & 127;
  const int x0   = (tile & 7) << 4;
  const int half = lane >> 4;
  const int mcol = lane & 15;

  const bf16_t* aBase = act + ppix(b, y, x0 + mcol) + (half << 3);
  const bf16_t* bBase = wpack + (((size_t)(wv << 2) * KSTEPS3 * 32) + lane) * 16;

  v8f acc[4];
#pragma unroll
  for (int j = 0; j < 4; ++j)
#pragma unroll
    for (int r = 0; r < 8; ++r) acc[j][r] = 0.0f;

  auto A_at = [&](int kstep) -> v16bf {
    const int kpos = kstep >> 3;
    const int cb   = (kstep & 7) << 5;
    const int uo   = (((kpos / 3) * PW + (kpos % 3) - (PW + 1)) << 8) + cb;
    union { v16bf v; v8bf h[2]; } au;
    au.h[0] = *(const v8bf*)(aBase + uo);
    au.h[1] = *(const v8bf*)(aBase + uo + 16);
    return au.v;
  };
  auto B_at = [&](int kstep, int j) -> v16bf {
    return *(const v16bf*)(bBase + (size_t)j * (KSTEPS3 * 32 * 16) + (size_t)kstep * 512);
  };

  // software pipeline: fetch kstep+1 while matrix pipe chews on kstep
  v16bf aC = A_at(0);
  v16bf b0 = B_at(0, 0), b1 = B_at(0, 1), b2 = B_at(0, 2), b3 = B_at(0, 3);
  for (int ks = 0; ks < KSTEPS3 - 1; ++ks) {
    const v16bf aN = A_at(ks + 1);
    const v16bf n0 = B_at(ks + 1, 0);
    const v16bf n1 = B_at(ks + 1, 1);
    const v16bf n2 = B_at(ks + 1, 2);
    const v16bf n3 = B_at(ks + 1, 3);
    acc[0] = wmma_bf16(aC, b0, acc[0]);
    acc[1] = wmma_bf16(aC, b1, acc[1]);
    acc[2] = wmma_bf16(aC, b2, acc[2]);
    acc[3] = wmma_bf16(aC, b3, acc[3]);
    aC = aN; b0 = n0; b1 = n1; b2 = n2; b3 = n3;
  }
  acc[0] = wmma_bf16(aC, b0, acc[0]);
  acc[1] = wmma_bf16(aC, b1, acc[1]);
  acc[2] = wmma_bf16(aC, b2, acc[2]);
  acc[3] = wmma_bf16(aC, b3, acc[3]);

#pragma unroll
  for (int j = 0; j < 4; ++j) {
    const int oc = (wv << 6) + (j << 4) + mcol;
    const float bi = bias[oc];
#pragma unroll
    for (int r = 0; r < 8; ++r) {
      float v = fmaxf(acc[j][r] + bi, 0.0f);
      const int xo = x0 + r + (half << 3);      // D row m = r + half*8
      out[ppix(b, y, xo) + oc] = (bf16_t)v;
    }
  }
}

// ---------------------------------------------------------------------------
// 1x1 conv heads: padded NHWC bf16 -> f32 (NCHW or [p][Cout]); bias, no ReLU
// ---------------------------------------------------------------------------
__global__ __launch_bounds__(64) void k_conv1x1(const bf16_t* __restrict__ act,
                                                const bf16_t* __restrict__ wpack,
                                                const float*  __restrict__ bias,
                                                float* __restrict__ out,
                                                int Cout, int ntiles, int nchw) {
  const int lane = threadIdx.x & 31;
  const int wv   = threadIdx.x >> 5;
  const int tile = blockIdx.x;
  const int b    = tile >> 10;
  const int y    = (tile >> 3) & 127;
  const int x0   = (tile & 7) << 4;
  const int half = lane >> 4;
  const int mcol = lane & 15;

  int myn[3] = {0, 0, 0};
  int cnt = 0;
  for (int nt = wv; nt < ntiles; nt += 2) myn[cnt++] = nt;

  v8f acc[3];
#pragma unroll
  for (int t = 0; t < 3; ++t)
#pragma unroll
    for (int r = 0; r < 8; ++r) acc[t][r] = 0.0f;

  const bf16_t* aBase = act + ppix(b, y, x0 + mcol) + (half << 3);
  for (int kstep = 0; kstep < KSTEPS1; ++kstep) {
    union { v16bf v; v8bf h[2]; } au;
    au.h[0] = *(const v8bf*)(aBase + (kstep << 5));
    au.h[1] = *(const v8bf*)(aBase + (kstep << 5) + 16);
#pragma unroll
    for (int t = 0; t < 3; ++t) {
      if (t < cnt) {
        const v16bf bfrag = *(const v16bf*)(wpack +
            ((((size_t)myn[t] * KSTEPS1 + kstep) << 5) + lane) * 16);
        acc[t] = wmma_bf16(au.v, bfrag, acc[t]);
      }
    }
  }

  for (int t = 0; t < cnt; ++t) {
    const int oc = myn[t] * 16 + mcol;
    if (oc >= Cout) continue;
    const float bi = bias[oc];
#pragma unroll
    for (int r = 0; r < 8; ++r) {
      const float v = acc[t][r] + bi;
      const int x = x0 + r + (half << 3);
      if (nchw) {
        out[((size_t)(b * Cout + oc)) * HWP + (size_t)y * WW + x] = v;
      } else {
        out[((size_t)b * HWP + (size_t)y * WW + x) * Cout + oc] = v;
      }
    }
  }
}

// ---------------------------------------------------------------------------
// Deformable conv: cooperative LDS-staged bilinear sampling + implicit GEMM.
// Per kpos: each of the 4 waves blends 2 of 8 channel-slices of the shared
// 16px x 256ch sampled A-tile into LDS (double-buffered), one barrier, then
// all waves run 8 ksteps of WMMA with A fragments fetched via ds_load_b128.
// ---------------------------------------------------------------------------
template <bool MASKF>
__global__ __launch_bounds__(128) void k_dcn(const bf16_t* __restrict__ feat,
                                             const float*  __restrict__ offs,
                                             const bf16_t* __restrict__ wpack,
                                             bf16_t* __restrict__ out) {
  __shared__ bf16_t atile[2][8][16][32];   // [buf][cs][px][ch]  = 16 KB

  const int lane = threadIdx.x & 31;
  const int wv   = threadIdx.x >> 5;
  const int tile = blockIdx.x;
  const int b    = tile >> 10;
  const int y    = (tile >> 3) & 127;
  const int x0   = (tile & 7) << 4;
  const int half = lane >> 4;
  const int mcol = lane & 15;
  const int x    = x0 + mcol;
  const size_t p = (size_t)b * HWP + (size_t)y * WW + x;

  v8f acc[4];
#pragma unroll
  for (int j = 0; j < 4; ++j)
#pragma unroll
    for (int r = 0; r < 8; ++r) acc[j][r] = 0.0f;

  const float* op = offs + p * 27;
  const bf16_t* bBase = wpack + (((size_t)(wv << 2) * KSTEPS3 * 32) + lane) * 16;

  for (int kpos = 0; kpos < 9; ++kpos) {
    // ---- bilinear sampling state for this lane's pixel -----------------
    const float offy = op[2 * kpos];
    const float offx = op[2 * kpos + 1];
    const float mval = MASKF ? op[18 + kpos] : 1.0f;
    const float py = (float)y + (float)(kpos / 3 - 1) + offy;
    const float px = (float)x + (float)(kpos % 3 - 1) + offx;
    const float fy = floorf(py), fx = floorf(px);
    const float ly = py - fy,   lx = px - fx;
    const int iy0 = (int)fy, ix0 = (int)fx;
    const int iy1 = iy0 + 1, ix1 = ix0 + 1;
    const float v00 = ((unsigned)iy0 < (unsigned)HH && (unsigned)ix0 < (unsigned)WW) ? 1.f : 0.f;
    const float v01 = ((unsigned)iy0 < (unsigned)HH && (unsigned)ix1 < (unsigned)WW) ? 1.f : 0.f;
    const float v10 = ((unsigned)iy1 < (unsigned)HH && (unsigned)ix0 < (unsigned)WW) ? 1.f : 0.f;
    const float v11 = ((unsigned)iy1 < (unsigned)HH && (unsigned)ix1 < (unsigned)WW) ? 1.f : 0.f;
    const float w00 = (1.f - ly) * (1.f - lx) * v00 * mval;
    const float w01 = (1.f - ly) * lx        * v01 * mval;
    const float w10 = ly        * (1.f - lx) * v10 * mval;
    const float w11 = ly        * lx         * v11 * mval;
    const int cy0 = min(max(iy0, 0), HH - 1), cx0 = min(max(ix0, 0), WW - 1);
    const int cy1 = min(max(iy1, 0), HH - 1), cx1 = min(max(ix1, 0), WW - 1);
    const size_t b00 = ppix(b, cy0, cx0), b01 = ppix(b, cy0, cx1);
    const size_t b10 = ppix(b, cy1, cx0), b11 = ppix(b, cy1, cx1);

    const int buf = kpos & 1;

    // ---- cooperative build: this wave fills cs slices 2*wv, 2*wv+1 -----
    const int ch0 = half << 4;                 // lanes 0-15 -> ch 0..15, 16-31 -> 16..31
#pragma unroll
    for (int s = 0; s < 2; ++s) {
      const int cs = (wv << 1) + s;
      const int cb = cs << 5;
#pragma unroll
      for (int run = 0; run < 2; ++run) {
        const int ch = cb + ch0 + (run << 3);
        const v8bf f00 = *(const v8bf*)(feat + b00 + ch);
        const v8bf f01 = *(const v8bf*)(feat + b01 + ch);
        const v8bf f10 = *(const v8bf*)(feat + b10 + ch);
        const v8bf f11 = *(const v8bf*)(feat + b11 + ch);
        v8bf rblend;
#pragma unroll
        for (int e = 0; e < 8; ++e) {
          const float fv = w00 * (float)f00[e] + w01 * (float)f01[e]
                         + w10 * (float)f10[e] + w11 * (float)f11[e];
          rblend[e] = (__bf16)fv;
        }
        *(v8bf*)&atile[buf][cs][mcol][ch0 + (run << 3)] = rblend;
      }
    }
    __syncthreads();   // A-tile for this kpos visible; prior buf reads done

    // ---- consume: 8 ksteps of WMMA, A fragments from LDS ---------------
    for (int cs = 0; cs < 8; ++cs) {
      union { v16bf v; v8bf h[2]; } au;
      au.h[0] = *(const v8bf*)&atile[buf][cs][mcol][half << 3];
      au.h[1] = *(const v8bf*)&atile[buf][cs][mcol][16 + (half << 3)];
      const int kstep = kpos * 8 + cs;
#pragma unroll
      for (int j = 0; j < 4; ++j) {
        const v16bf bfrag = *(const v16bf*)(bBase +
            (size_t)j * (KSTEPS3 * 32 * 16) + (size_t)kstep * 512);
        acc[j] = wmma_bf16(au.v, bfrag, acc[j]);
      }
    }
  }

#pragma unroll
  for (int j = 0; j < 4; ++j) {
    const int oc = (wv << 6) + (j << 4) + mcol;
#pragma unroll
    for (int r = 0; r < 8; ++r) {
      float v = fmaxf(acc[j][r], 0.0f);         // no bias on deform conv
      const int xo = x0 + r + (half << 3);
      out[ppix(b, y, xo) + oc] = (bf16_t)v;
    }
  }
}

// ---------------------------------------------------------------------------
// Final stats: refine = ref_raw + offset ; mean over 9 pts ; wh = max |pt+mean|
// ---------------------------------------------------------------------------
__global__ void k_stats(const float* __restrict__ ref_raw,
                        const float* __restrict__ offs,
                        float* __restrict__ out) {
  const int p = blockIdx.x * blockDim.x + threadIdx.x;
  if (p >= PTOT) return;
  const int b  = p >> 14;
  const int yx = p & 16383;
  float r[18];
  float m0 = 0.f, m1 = 0.f;
#pragma unroll
  for (int i = 0; i < 9; ++i) {
    r[2 * i]     = ref_raw[(size_t)p * 18 + 2 * i]     + offs[(size_t)p * 27 + 2 * i];
    r[2 * i + 1] = ref_raw[(size_t)p * 18 + 2 * i + 1] + offs[(size_t)p * 27 + 2 * i + 1];
    m0 += r[2 * i];
    m1 += r[2 * i + 1];
  }
  m0 *= (1.0f / 9.0f);
  m1 *= (1.0f / 9.0f);
  float wh0 = 0.f, wh1 = 0.f;
#pragma unroll
  for (int i = 0; i < 9; ++i) {
    wh0 = fmaxf(wh0, fabsf(r[2 * i]     + m0));
    wh1 = fmaxf(wh1, fabsf(r[2 * i + 1] + m1));
  }
  const size_t cls_sz = (size_t)BB * 80 * HWP;
  out[cls_sz + ((size_t)(b * 2 + 0)) * HWP + yx] = wh0;
  out[cls_sz + ((size_t)(b * 2 + 1)) * HWP + yx] = wh1;
  const size_t rbase = cls_sz + (size_t)BB * 2 * HWP;
  out[rbase + ((size_t)(b * 2 + 0)) * HWP + yx] = m0;
  out[rbase + ((size_t)(b * 2 + 1)) * HWP + yx] = m1;
}

// ---------------------------------------------------------------------------
extern "C" void kernel_launch(void* const* d_in, const int* in_sizes, int n_in,
                              void* d_out, int out_size, void* d_ws, size_t ws_size,
                              hipStream_t stream) {
  const float* x           = (const float*)d_in[0];
  const float* cls_w[3]    = {(const float*)d_in[1], (const float*)d_in[5], (const float*)d_in[9]};
  const float* cls_b[3]    = {(const float*)d_in[2], (const float*)d_in[6], (const float*)d_in[10]};
  const float* reg_w[3]    = {(const float*)d_in[3], (const float*)d_in[7], (const float*)d_in[11]};
  const float* reg_b[3]    = {(const float*)d_in[4], (const float*)d_in[8], (const float*)d_in[12]};
  const float* init_conv_w = (const float*)d_in[13];
  const float* init_conv_b = (const float*)d_in[14];
  const float* init_out_w  = (const float*)d_in[15];
  const float* init_out_b  = (const float*)d_in[16];
  const float* dcn_cls_w   = (const float*)d_in[17];
  const float* cls_out_w   = (const float*)d_in[18];
  const float* cls_out_b   = (const float*)d_in[19];
  const float* dcn_ref_w   = (const float*)d_in[20];
  const float* ref_out_w   = (const float*)d_in[21];
  const float* ref_out_b   = (const float*)d_in[22];

  char* ws = (char*)d_ws;
  size_t off = 0;
  auto carve = [&](size_t bytes) -> void* {
    void* ptr = ws + off;
    off += (bytes + 255) & ~(size_t)255;
    return ptr;
  };
  const size_t ACTP = (size_t)BB * PIXP * CC * sizeof(bf16_t);   // padded act buffer
  const size_t W3P  = (size_t)16 * KSTEPS3 * 32 * 16 * sizeof(bf16_t);

  // five padded activation buffers, carved contiguously so one zero pass covers all
  bf16_t* xb   = (bf16_t*)carve(ACTP);
  bf16_t* tmpA = (bf16_t*)carve(ACTP);
  bf16_t* tmpB = (bf16_t*)carve(ACTP);
  bf16_t* clsf = (bf16_t*)carve(ACTP);
  bf16_t* ptsf = (bf16_t*)carve(ACTP);
  float*  offsb  = (float*)carve((size_t)PTOT * 27 * sizeof(float));
  float*  refraw = (float*)carve((size_t)PTOT * 18 * sizeof(float));
  bf16_t* wp3[9];
  for (int i = 0; i < 9; ++i) wp3[i] = (bf16_t*)carve(W3P);
  bf16_t* wp_io  = (bf16_t*)carve((size_t)2 * KSTEPS1 * 32 * 16 * sizeof(bf16_t));
  bf16_t* wp_co  = (bf16_t*)carve((size_t)5 * KSTEPS1 * 32 * 16 * sizeof(bf16_t));
  bf16_t* wp_ro  = (bf16_t*)carve((size_t)2 * KSTEPS1 * 32 * 16 * sizeof(bf16_t));

  // ---- zero padded activation region (halo stays zero; interiors rewritten)
  {
    const size_t n16 = (5 * ACTP) / 16;     // ACTP is 256B-aligned size
    k_zero16<<<(int)((n16 + 255) / 256), 256, 0, stream>>>((v4u*)xb, n16);
  }

  // ---- weight packing --------------------------------------------------
  const float* w3src[9] = {cls_w[0], cls_w[1], cls_w[2],
                           reg_w[0], reg_w[1], reg_w[2],
                           init_conv_w, dcn_cls_w, dcn_ref_w};
  {
    const size_t n = (size_t)16 * KSTEPS3 * 32 * 16;
    const int blocks = (int)((n + 255) / 256);
    for (int i = 0; i < 9; ++i)
      k_pack_w3x3<<<blocks, 256, 0, stream>>>(w3src[i], wp3[i]);
  }
  {
    auto launch1 = [&](const float* w, bf16_t* dst, int O, int nt) {
      const size_t n = (size_t)nt * KSTEPS1 * 32 * 16;
      k_pack_w1x1<<<(int)((n + 255) / 256), 256, 0, stream>>>(w, dst, O, nt);
    };
    launch1(init_out_w, wp_io, 27, 2);
    launch1(cls_out_w,  wp_co, 80, 5);
    launch1(ref_out_w,  wp_ro, 18, 2);
  }

  // ---- activation conversion ------------------------------------------
  {
    const size_t n = (size_t)PTOT * CC;
    k_nchw_to_nhwc<<<(int)((n + 255) / 256), 256, 0, stream>>>(x, xb);
  }

  // ---- towers ----------------------------------------------------------
  k_conv3x3<<<MTILES, 128, 0, stream>>>(xb,   wp3[0], cls_b[0], tmpA);
  k_conv3x3<<<MTILES, 128, 0, stream>>>(tmpA, wp3[1], cls_b[1], tmpB);
  k_conv3x3<<<MTILES, 128, 0, stream>>>(tmpB, wp3[2], cls_b[2], clsf);
  k_conv3x3<<<MTILES, 128, 0, stream>>>(xb,   wp3[3], reg_b[0], tmpA);
  k_conv3x3<<<MTILES, 128, 0, stream>>>(tmpA, wp3[4], reg_b[1], tmpB);
  k_conv3x3<<<MTILES, 128, 0, stream>>>(tmpB, wp3[5], reg_b[2], ptsf);

  // ---- init branch: offsets + mask ------------------------------------
  k_conv3x3<<<MTILES, 128, 0, stream>>>(ptsf, wp3[6], init_conv_b, tmpA);
  k_conv1x1<<<MTILES, 64, 0, stream>>>(tmpA, wp_io, init_out_b, offsb, 27, 2, 0);

  // ---- deformable conv branches ---------------------------------------
  k_dcn<true><<<MTILES, 128, 0, stream>>>(clsf, offsb, wp3[7], tmpB);
  k_conv1x1<<<MTILES, 64, 0, stream>>>(tmpB, wp_co, cls_out_b, (float*)d_out, 80, 5, 1);

  k_dcn<false><<<MTILES, 128, 0, stream>>>(ptsf, offsb, wp3[8], tmpA);
  k_conv1x1<<<MTILES, 64, 0, stream>>>(tmpA, wp_ro, ref_out_b, refraw, 18, 2, 0);

  // ---- final stats -----------------------------------------------------
  k_stats<<<(PTOT + 255) / 256, 256, 0, stream>>>(refraw, offsb, (float*)d_out);
}
